// NeuralCDE_51264729645500
// MI455X (gfx1250) — matrix operand order repaired and checked by reading, not compiled, not run on hardware
//
#include <hip/hip_runtime.h>
#include <hip/hip_bf16.h>
#include <math.h>

// ---------------- problem constants ----------------
#define HID     128
#define DATA    33
#define VFW     512
#define NB      16
#define NPTS    200
#define NSTEPS  250
#define L3OUT   (HID*DATA)        // 4224
#define DTSTEP  (1.0f/250.0f)

// ---------------- parallel decomposition ----------------
#define NW        32              // workgroups (one per WGP)
#define TPB       256             // 8 wave32 per WG
#define SL_SMALL  16              // out-cols per WG for L0/L1/L2 (16*32 = 512)
#define HID_PER_WG 4              // 128/32 hidden rows owned per WG
#define L3_OWN    (HID_PER_WG*DATA)  // 132 owned L3 columns
#define L3_TILES  9               // ceil(132/16), covers 144 cols
#define L3_ROWS   144             // W3 rows staged in LDS per WG

typedef __bf16 bf16_t;
typedef bf16_t v16bf __attribute__((ext_vector_type(16)));
typedef bf16_t v8bf  __attribute__((ext_vector_type(8)));
typedef float  v8f   __attribute__((ext_vector_type(8)));

// ---------- CDNA5 async global->LDS copy (ASYNCcnt tracked) ----------
__device__ __forceinline__ void async_ld16(void* lds_dst, const void* gsrc) {
  unsigned l = (unsigned)(uintptr_t)lds_dst;   // low 32 bits of a shared ptr = LDS offset
  asm volatile("global_load_async_to_lds_b128 %0, %1, off"
               :: "v"(l), "v"(gsrc) : "memory");
}
__device__ __forceinline__ void wait_async0() {
  asm volatile("s_wait_asynccnt 0x0" ::: "memory");
}

// ---------- WMMA fragment loaders (per CDNA5 ISA VGPR layouts) ----------
// A tile: 16 rows (batch) x K=32, source row-major [16][kstride]
__device__ __forceinline__ v16bf load_a_frag(const bf16_t* base, int kstride,
                                             int kbase, int lane) {
  int row  = lane & 15;
  int half = lane >> 4;
  const bf16_t* p = base + (size_t)row * kstride + kbase + half * 8;
  v8bf lo = *(const v8bf*)(p);          // K = kbase + half*8 + 0..7
  v8bf hi = *(const v8bf*)(p + 16);     // K = kbase + 16 + half*8 + 0..7
  return __builtin_shufflevector(lo, hi, 0,1,2,3,4,5,6,7,8,9,10,11,12,13,14,15);
}
// B tile: K=32 x 16 cols; weights stored row-major [out][kstride] so each lane
// reads K = half*8 + 0..7 and 16 + half*8 + 0..7 of its output row.
__device__ __forceinline__ v16bf load_b_frag(const bf16_t* w, int kstride,
                                             int kbase, int lane) {
  int n    = lane & 15;
  int half = lane >> 4;
  const bf16_t* p = w + (size_t)n * kstride + kbase + half * 8;
  v8bf lo = *(const v8bf*)(p);
  v8bf hi = *(const v8bf*)(p + 16);
  return __builtin_shufflevector(lo, hi, 0,1,2,3,4,5,6,7,8,9,10,11,12,13,14,15);
}
__device__ __forceinline__ v8f wmma_bf16(v16bf a, v16bf b, v8f c) {
  return __builtin_amdgcn_wmma_f32_16x16x32_bf16(false, a, false, b,
                                                 (short)0, c, false, false);
}

// ---------- fast activations (v_exp_f32 + v_rcp_f32, no IEEE divide) ----------
__device__ __forceinline__ float silu_f(float v) {
  return v * __builtin_amdgcn_rcpf(1.0f + __expf(-v));
}
__device__ __forceinline__ float tanh_f(float v) {
  return 1.0f - 2.0f * __builtin_amdgcn_rcpf(1.0f + __expf(2.0f * v));
}
__device__ __forceinline__ float sigmoid_f(float v) {
  return __builtin_amdgcn_rcpf(1.0f + __expf(-v));
}

// ---------- sense-reversing device-scope barrier across NW workgroups ----------
__device__ __forceinline__ void gbar(int* cnt, int* gen) {
  __builtin_amdgcn_fence(__ATOMIC_RELEASE, "agent");   // write-back producer data
  __syncthreads();
  if (threadIdx.x == 0) {
    int g = __hip_atomic_load(gen, __ATOMIC_RELAXED, __HIP_MEMORY_SCOPE_AGENT);
    int v = __hip_atomic_fetch_add(cnt, 1, __ATOMIC_RELAXED, __HIP_MEMORY_SCOPE_AGENT);
    if (v == NW - 1) {
      __hip_atomic_store(cnt, 0, __ATOMIC_RELAXED, __HIP_MEMORY_SCOPE_AGENT);
      __hip_atomic_store(gen, g + 1, __ATOMIC_RELEASE, __HIP_MEMORY_SCOPE_AGENT);
    } else {
      while (__hip_atomic_load(gen, __ATOMIC_RELAXED, __HIP_MEMORY_SCOPE_AGENT) == g)
        __builtin_amdgcn_s_sleep(1);
    }
  }
  __syncthreads();
  __builtin_amdgcn_fence(__ATOMIC_ACQUIRE, "agent");   // invalidate stale lines for all waves
}

// =================== prep kernel 1: f32 -> bf16 weights ===================
__global__ void k_convert(const float* __restrict__ w0, const float* __restrict__ w1,
                          const float* __restrict__ w2, const float* __restrict__ w3,
                          bf16_t* Wb0, bf16_t* Wb1, bf16_t* Wb2, bf16_t* Wb3,
                          int* bar_cnt, int* bar_gen) {
  size_t i = (size_t)blockIdx.x * blockDim.x + threadIdx.x;
  if (i == 0) { *bar_cnt = 0; *bar_gen = 0; }
  const size_t n0 = (size_t)VFW * HID;
  const size_t n1 = (size_t)VFW * VFW;
  const size_t n3 = (size_t)L3OUT * VFW;
  if (i < n0)                  Wb0[i]             = (bf16_t)w0[i];
  else if (i < n0 + n1)        Wb1[i - n0]        = (bf16_t)w1[i - n0];
  else if (i < n0 + 2*n1)      Wb2[i - n0 - n1]   = (bf16_t)w2[i - n0 - n1];
  else if (i < n0 + 2*n1 + n3) Wb3[i - n0 - 2*n1] = (bf16_t)w3[i - n0 - 2*n1];
}

// ===== prep kernel 2: Hermite ctrl_deriv table Z[251][16][33] and y0 =====
__global__ void k_prep(const float* __restrict__ X, const float* __restrict__ l1w,
                       const float* __restrict__ l1b, float* __restrict__ Zt,
                       float* __restrict__ y0) {
  const int NZT = (NSTEPS + 1) * NB * DATA;
  int i = blockIdx.x * blockDim.x + threadIdx.x;
  if (i < NZT) {
    int tidx = i / (NB * DATA);
    int rem  = i % (NB * DATA);
    int b = rem / DATA, d = rem % DATA;
    const float* Xb = X + (size_t)b * NPTS * DATA;
    float t = (float)tidx * DTSTEP;
    // searchsorted(ts, t, 'right') - 1, clipped to [0, NPTS-2]
    int lo = 0, hi = NPTS - 2;
    while (lo < hi) {
      int mid = (lo + hi + 1) >> 1;
      if (Xb[(size_t)mid * DATA] <= t) lo = mid; else hi = mid - 1;
    }
    int seg = lo;
    float t0 = Xb[(size_t)seg * DATA];
    float t1 = Xb[(size_t)(seg + 1) * DATA];
    float dt = t1 - t0;
    float xv = t - t0;
    float delta = (Xb[(size_t)(seg+1)*DATA + d] - Xb[(size_t)seg*DATA + d]) / dt;
    float f0 = 0.0f;
    if (seg > 0) {
      float tp = Xb[(size_t)(seg-1)*DATA];
      f0 = (Xb[(size_t)seg*DATA + d] - Xb[(size_t)(seg-1)*DATA + d]) / (t0 - tp);
    }
    float cc = 2.0f * (delta - f0) / dt;
    float dd = (f0 - delta) / (dt * dt);
    Zt[i] = f0 + xv * (2.0f * cc + 3.0f * dd * xv);   // b + x(2c + 3dx)
  } else {
    int j = i - NZT;
    if (j < NB * HID) {
      int b = j / HID, h = j % HID;
      const float* Xb = X + (size_t)b * NPTS * DATA;
      float s = l1b[h];
      for (int d = 0; d < DATA; ++d) s += l1w[h * DATA + d] * Xb[d];
      y0[j] = s;   // h0 = l1_w @ ctrl_eval(0) + l1_b  (ctrl_eval(0) = X[:,0,:])
    }
  }
}

// =================== persistent main kernel ===================
__global__ __launch_bounds__(TPB, 1)
void ncde_main(const bf16_t* __restrict__ Wb0, const bf16_t* __restrict__ Wb1,
               const bf16_t* __restrict__ Wb2, const bf16_t* __restrict__ Wb3,
               const float* __restrict__ b0, const float* __restrict__ b1,
               const float* __restrict__ b2, const float* __restrict__ b3,
               const float* __restrict__ Zt, const float* __restrict__ y0,
               bf16_t* hbuf0, bf16_t* hbuf1, bf16_t* hbuf2,
               float* kbuf0, float* kbuf1,
               int* bar_cnt, int* bar_gen,
               const float* __restrict__ l2w, const float* __restrict__ l2b,
               float* __restrict__ out) {
  // -------- LDS (~220 KB of the WGP's 320 KB) --------
  __shared__ __attribute__((aligned(16))) bf16_t sW3[L3_ROWS * VFW];   // 144 KB
  __shared__ __attribute__((aligned(16))) bf16_t sW1[SL_SMALL * VFW];  // 16 KB
  __shared__ __attribute__((aligned(16))) bf16_t sW2[SL_SMALL * VFW];  // 16 KB
  __shared__ __attribute__((aligned(16))) bf16_t sW0[SL_SMALL * HID];  // 4 KB
  __shared__ __attribute__((aligned(16))) bf16_t sH [NB * VFW];        // 16 KB act staging
  __shared__ __attribute__((aligned(16))) bf16_t sYb[NB * HID];        // 4 KB  y bf16 (A of L0)
  __shared__ float sY  [NB * HID];        // y (f32 state)
  __shared__ float sK1 [NB * HID];        // k1 of current Heun step
  __shared__ float sZ  [NB * DATA];       // ctrl_deriv(t) for this phase
  __shared__ float sAcc[NB * SL_SMALL];   // K-split reduction tile
  __shared__ float sKacc[NB * HID_PER_WG];
  __shared__ float sB0[SL_SMALL], sB1[SL_SMALL], sB2[SL_SMALL], sB3[L3_ROWS];

  const int wg   = blockIdx.x;
  const int tid  = threadIdx.x;
  const int lane = tid & 31;
  const int wv   = tid >> 5;

  // -------- one-time: pin weight slices in LDS via async DMA --------
  for (int i = tid; i < L3_ROWS * VFW / 8; i += TPB) {
    int r = i / (VFW / 8), c8 = i % (VFW / 8);
    int grow = L3_OWN * wg + r;
    if (grow > L3OUT - 1) grow = L3OUT - 1;   // clamp (cols >= 4224 are never consumed)
    async_ld16(sW3 + (size_t)r * VFW + c8 * 8, Wb3 + (size_t)grow * VFW + c8 * 8);
  }
  for (int i = tid; i < SL_SMALL * VFW / 8; i += TPB) {
    int r = i / (VFW / 8), c8 = i % (VFW / 8);
    async_ld16(sW1 + (size_t)r * VFW + c8 * 8, Wb1 + (size_t)(SL_SMALL*wg + r) * VFW + c8 * 8);
    async_ld16(sW2 + (size_t)r * VFW + c8 * 8, Wb2 + (size_t)(SL_SMALL*wg + r) * VFW + c8 * 8);
  }
  for (int i = tid; i < SL_SMALL * HID / 8; i += TPB) {
    int r = i / (HID / 8), c8 = i % (HID / 8);
    async_ld16(sW0 + (size_t)r * HID + c8 * 8, Wb0 + (size_t)(SL_SMALL*wg + r) * HID + c8 * 8);
  }
  if (tid < SL_SMALL) {
    sB0[tid] = b0[SL_SMALL*wg + tid];
    sB1[tid] = b1[SL_SMALL*wg + tid];
    sB2[tid] = b2[SL_SMALL*wg + tid];
  }
  if (tid < L3_ROWS) {
    int g = L3_OWN * wg + tid;
    sB3[tid] = (g < L3OUT) ? b3[g] : 0.0f;
  }
  for (int i = tid; i < NB * HID; i += TPB) { sY[i] = y0[i]; sYb[i] = (bf16_t)y0[i]; }
  wait_async0();
  __syncthreads();

  // -------- 250 Heun steps x 2 MLP evals --------
  #pragma unroll 1
  for (int step = 0; step < NSTEPS; ++step) {
    #pragma unroll 1
    for (int sub = 0; sub < 2; ++sub) {
      const int tix = step + sub;   // deriv table index for this eval
      float* kb = sub ? kbuf1 : kbuf0;   // parity double-buffer removes WAR barrier
      for (int i = tid; i < NB * DATA; i += TPB) sZ[i] = Zt[(size_t)tix * NB * DATA + i];
      sAcc[tid] = 0.0f;
      if (tid < NB * HID_PER_WG) sKacc[tid] = 0.0f;
      __syncthreads();

      // ---- L0: (16x128) @ W0^T slice -> 16x16, K split over 4 waves ----
      if (wv < 4) {
        v8f acc = {};
        int kbase = wv * 32;
        acc = wmma_bf16(load_a_frag(sYb, HID, kbase, lane),
                        load_b_frag(sW0, HID, kbase, lane), acc);
        int half = lane >> 4, n = lane & 15;
        #pragma unroll
        for (int j = 0; j < 8; ++j) atomicAdd(&sAcc[(j + half*8) * SL_SMALL + n], acc[j]);
      }
      __syncthreads();
      if (tid < NB * SL_SMALL / 2) {    // packed 2xbf16 dword stores
        int m = tid >> 3, n2 = (tid & 7) * 2;
        union { bf16_t h[2]; unsigned u; } pk;
        pk.h[0] = (bf16_t)silu_f(sAcc[m*SL_SMALL + n2]     + sB0[n2]);
        pk.h[1] = (bf16_t)silu_f(sAcc[m*SL_SMALL + n2 + 1] + sB0[n2+1]);
        *(unsigned*)(hbuf0 + (size_t)m * VFW + SL_SMALL*wg + n2) = pk.u;
      }
      gbar(bar_cnt, bar_gen);

      // ---- L1: (16x512) @ W1^T slice, K split over 8 waves ----
      for (int i = tid; i < NB * VFW / 8; i += TPB)
        async_ld16(sH + i * 8, hbuf0 + i * 8);
      sAcc[tid] = 0.0f;
      wait_async0();
      __syncthreads();
      {
        v8f acc = {};
        #pragma unroll
        for (int kk = 0; kk < 2; ++kk) {
          int kbase = (wv * 2 + kk) * 32;
          acc = wmma_bf16(load_a_frag(sH, VFW, kbase, lane),
                          load_b_frag(sW1, VFW, kbase, lane), acc);
        }
        int half = lane >> 4, n = lane & 15;
        #pragma unroll
        for (int j = 0; j < 8; ++j) atomicAdd(&sAcc[(j + half*8) * SL_SMALL + n], acc[j]);
      }
      __syncthreads();
      if (tid < NB * SL_SMALL / 2) {
        int m = tid >> 3, n2 = (tid & 7) * 2;
        union { bf16_t h[2]; unsigned u; } pk;
        pk.h[0] = (bf16_t)silu_f(sAcc[m*SL_SMALL + n2]     + sB1[n2]);
        pk.h[1] = (bf16_t)silu_f(sAcc[m*SL_SMALL + n2 + 1] + sB1[n2+1]);
        *(unsigned*)(hbuf1 + (size_t)m * VFW + SL_SMALL*wg + n2) = pk.u;
      }
      gbar(bar_cnt, bar_gen);

      // ---- L2 ----
      for (int i = tid; i < NB * VFW / 8; i += TPB)
        async_ld16(sH + i * 8, hbuf1 + i * 8);
      sAcc[tid] = 0.0f;
      wait_async0();
      __syncthreads();
      {
        v8f acc = {};
        #pragma unroll
        for (int kk = 0; kk < 2; ++kk) {
          int kbase = (wv * 2 + kk) * 32;
          acc = wmma_bf16(load_a_frag(sH, VFW, kbase, lane),
                          load_b_frag(sW2, VFW, kbase, lane), acc);
        }
        int half = lane >> 4, n = lane & 15;
        #pragma unroll
        for (int j = 0; j < 8; ++j) atomicAdd(&sAcc[(j + half*8) * SL_SMALL + n], acc[j]);
      }
      __syncthreads();
      if (tid < NB * SL_SMALL / 2) {
        int m = tid >> 3, n2 = (tid & 7) * 2;
        union { bf16_t h[2]; unsigned u; } pk;
        pk.h[0] = (bf16_t)silu_f(sAcc[m*SL_SMALL + n2]     + sB2[n2]);
        pk.h[1] = (bf16_t)silu_f(sAcc[m*SL_SMALL + n2 + 1] + sB2[n2+1]);
        *(unsigned*)(hbuf2 + (size_t)m * VFW + SL_SMALL*wg + n2) = pk.u;
      }
      gbar(bar_cnt, bar_gen);

      // ---- L3 (16x512 @ 144-col slice) + tanh + contraction over DATA ----
      for (int i = tid; i < NB * VFW / 8; i += TPB)
        async_ld16(sH + i * 8, hbuf2 + i * 8);
      wait_async0();
      __syncthreads();
      #pragma unroll 1
      for (int t = wv; t < L3_TILES; t += 8) {
        v8f acc = {};
        for (int ks = 0; ks < 16; ++ks)
          acc = wmma_bf16(load_a_frag(sH, VFW, ks * 32, lane),
                          load_b_frag(sW3 + (size_t)(t * 16) * VFW, VFW, ks * 32, lane), acc);
        int nl = t * 16 + (lane & 15);       // local out column
        int ng = L3_OWN * wg + nl;           // global out column
        if (nl < L3_OWN && ng < L3OUT) {
          int half = lane >> 4;
          int d  = nl % DATA;                // (132*wg) % 33 == 0
          int hl = nl / DATA;                // owned hid row, 0..3
          #pragma unroll
          for (int j = 0; j < 8; ++j) {
            int m = j + half * 8;            // batch row
            float v = tanh_f(acc[j] + sB3[nl]);
            atomicAdd(&sKacc[m * HID_PER_WG + hl], v * sZ[m * DATA + d]);
          }
        }
      }
      __syncthreads();
      if (tid < NB * HID_PER_WG) {
        int m = tid / HID_PER_WG, hl = tid % HID_PER_WG;
        kb[(size_t)m * HID + HID_PER_WG*wg + hl] = sKacc[tid];
      }
      gbar(bar_cnt, bar_gen);   // k-slices visible

      // ---- Heun state update (every WG keeps full y locally) ----
      for (int i = tid; i < NB * HID; i += TPB) {
        float kv = kb[i];
        if (sub == 0) {
          sK1[i] = kv;
          float ym = sY[i] + DTSTEP * kv;       // y + dt*k1 for k2 eval
          sYb[i] = (bf16_t)ym;
        } else {
          float yn = sY[i] + 0.5f * DTSTEP * (sK1[i] + kv);
          sY[i]  = yn;
          sYb[i] = (bf16_t)yn;
        }
      }
      // no trailing barrier needed: kbuf parity double-buffering makes the next
      // writer of this buffer seven global barriers downstream.
    }
  }

  // -------- head: sigmoid(l2_w @ yT + l2_b), WG 0 writes the 16 outputs ----
  __syncthreads();
  if (wg == 0 && tid < NB) {
    float s = l2b[0];
    for (int h = 0; h < HID; ++h) s += l2w[h] * sY[tid * HID + h];
    out[tid] = sigmoid_f(s);
  }
}

// =================== host launcher ===================
extern "C" void kernel_launch(void* const* d_in, const int* in_sizes, int n_in,
                              void* d_out, int out_size, void* d_ws, size_t ws_size,
                              hipStream_t stream) {
  (void)in_sizes; (void)n_in; (void)out_size; (void)ws_size;
  const float* X    = (const float*)d_in[0];
  const float* w0   = (const float*)d_in[1];
  const float* vb0  = (const float*)d_in[2];
  const float* w1   = (const float*)d_in[3];
  const float* vb1  = (const float*)d_in[4];
  const float* w2   = (const float*)d_in[5];
  const float* vb2  = (const float*)d_in[6];
  const float* w3   = (const float*)d_in[7];
  const float* vb3  = (const float*)d_in[8];
  const float* l1w  = (const float*)d_in[9];
  const float* l1b  = (const float*)d_in[10];
  const float* l2w  = (const float*)d_in[11];
  const float* l2b  = (const float*)d_in[12];

  char* ws = (char*)d_ws;
  size_t off = 0;
  auto take = [&](size_t bytes) { size_t r = off; off = (off + bytes + 255) & ~(size_t)255; return r; };
  bf16_t* Wb0 = (bf16_t*)(ws + take((size_t)VFW * HID * 2));
  bf16_t* Wb1 = (bf16_t*)(ws + take((size_t)VFW * VFW * 2));
  bf16_t* Wb2 = (bf16_t*)(ws + take((size_t)VFW * VFW * 2));
  bf16_t* Wb3 = (bf16_t*)(ws + take((size_t)L3OUT * VFW * 2));
  float*  Zt  = (float*) (ws + take((size_t)(NSTEPS + 1) * NB * DATA * 4));
  float*  y0  = (float*) (ws + take((size_t)NB * HID * 4));
  bf16_t* h0b = (bf16_t*)(ws + take((size_t)NB * VFW * 2));
  bf16_t* h1b = (bf16_t*)(ws + take((size_t)NB * VFW * 2));
  bf16_t* h2b = (bf16_t*)(ws + take((size_t)NB * VFW * 2));
  float*  kb0 = (float*) (ws + take((size_t)NB * HID * 4));
  float*  kb1 = (float*) (ws + take((size_t)NB * HID * 4));
  int*    bar = (int*)   (ws + take(2 * sizeof(int)));

  const size_t nconv = (size_t)VFW*HID + 2*(size_t)VFW*VFW + (size_t)L3OUT*VFW;
  k_convert<<<(unsigned)((nconv + 255) / 256), 256, 0, stream>>>(
      w0, w1, w2, w3, Wb0, Wb1, Wb2, Wb3, bar, bar + 1);

  const int nprep = (NSTEPS + 1) * NB * DATA + NB * HID;
  k_prep<<<(nprep + 255) / 256, 256, 0, stream>>>(X, l1w, l1b, Zt, y0);

  ncde_main<<<NW, TPB, 0, stream>>>(Wb0, Wb1, Wb2, Wb3,
                                    vb0, vb1, vb2, vb3,
                                    Zt, y0, h0b, h1b, h2b, kb0, kb1,
                                    bar, bar + 1, l2w, l2b, (float*)d_out);
}
